// SpatialAttentionModule_76347338653968
// MI455X (gfx1250) — matrix operand order repaired and checked by reading, not compiled, not run on hardware
//
#include <hip/hip_runtime.h>
#include <hip/hip_bf16.h>
#include <math.h>

// ---------------------------------------------------------------------------
// SpatialAttentionModule for MI455X (gfx1250).
// bf16 WMMA everywhere + fused flash-style attention with double-buffered
// async LDS staging (global_load_async_to_lds_b128 / s_wait_asynccnt).
// B=8, C=256, H=W=64 -> N=4096, CQK=64.
// ---------------------------------------------------------------------------

typedef __attribute__((ext_vector_type(16))) __bf16 v16bf;
typedef __attribute__((ext_vector_type(8)))  __bf16 v8bf;
typedef __attribute__((ext_vector_type(8)))  float  v8f;

#define KB_N 4096
#define KB_C 256
#define KB_D 64
#define KB_B 8

static __device__ inline v8f wmma_bf16(v16bf a, v16bf b, v8f c) {
  return __builtin_amdgcn_wmma_f32_16x16x32_bf16(
      /*neg_a=*/false, a, /*neg_b=*/false, b,
      /*c_mod=*/(short)0, c, /*reuse_a=*/false, /*reuse_b=*/false);
}

// Load 16 bf16 per lane as two 16-byte chunks: elems[0..7] at p, elems[8..15]
// at p+gap.  gap==8  -> fully contiguous 16 values (not used in this file)
//            gap==16 -> A-operand per-lane slice (K groups 0..7 and 16..23)
static __device__ inline v16bf load16(const __bf16* p, int gap) {
  v8bf lo = *(const v8bf*)p;
  v8bf hi = *(const v8bf*)(p + gap);
  v16bf r;
#pragma unroll
  for (int i = 0; i < 8; ++i) { r[i] = lo[i]; r[8 + i] = hi[i]; }
  return r;
}

// One 16-byte async global->LDS copy (CDNA5 ASYNCcnt path).
static __device__ inline void async_b128(unsigned lds_off, const void* gaddr) {
  asm volatile("global_load_async_to_lds_b128 %0, %1, off"
               :: "v"(lds_off), "v"(gaddr)
               : "memory");
}

static __device__ inline unsigned lds_off32(const void* p) {
  // Low 32 bits of a generic pointer to LDS are the workgroup-relative offset.
  return (unsigned)(unsigned long long)p;
}

// ---------------------------------------------------------------------------
// Prep kernels: f32 -> bf16 converts (+ transpose of x to [b][n][c])
// ---------------------------------------------------------------------------
__global__ void sam_convert_transpose_x(const float* __restrict__ x,
                                        __bf16* __restrict__ xb) {
  size_t i = (size_t)blockIdx.x * blockDim.x + threadIdx.x;  // over b*n*c
  int c = (int)(i & (KB_C - 1));
  size_t bn = i >> 8;
  int n = (int)(bn & (KB_N - 1));
  int b = (int)(bn >> 12);
  xb[i] = (__bf16)x[((size_t)b * KB_C + c) * KB_N + n];
}

__global__ void sam_convert_bf16(const float* __restrict__ s,
                                 __bf16* __restrict__ d, int n) {
  int i = blockIdx.x * blockDim.x + threadIdx.x;
  if (i < n) d[i] = (__bf16)s[i];
}

// ---------------------------------------------------------------------------
// Projection GEMM: Y[b][n][o] = sum_c xb[b][n][c] * W[o][c] + bias[o]
// Each wave computes one 16x16 tile; block = 8 waves covering 128 rows.
// transposed==1 stores Y to y[b][o][n] instead (used for V: natural [C][N]).
// ---------------------------------------------------------------------------
__global__ __launch_bounds__(256) void sam_proj_gemm(
    const __bf16* __restrict__ xb, const __bf16* __restrict__ wbf,
    const float* __restrict__ bias, __bf16* __restrict__ y, int Oc,
    int transposed) {
  int b = blockIdx.y;
  int tilesPerRowBlk = Oc >> 4;
  int rowblk = blockIdx.x / tilesPerRowBlk;
  int colblk = blockIdx.x % tilesPerRowBlk;
  int wave = threadIdx.x >> 5;
  int lane = threadIdx.x & 31;
  int half = lane >> 4;
  int lr = lane & 15;
  int r0 = rowblk * 128 + wave * 16;
  int c0 = colblk * 16;

  const __bf16* xrow = xb + ((size_t)b * KB_N + r0 + lr) * KB_C;
  const __bf16* wrow = wbf + (size_t)(c0 + lr) * KB_C;

  v8f acc = {};
#pragma unroll
  for (int kk = 0; kk < KB_C; kk += 32) {
    v16bf a = load16(xrow + kk + half * 8, 16);
    v16bf bm = load16(wrow + kk + half * 8, 16);
    acc = wmma_bf16(a, bm, acc);
  }

  int o = c0 + lr;
  float bv = bias[o];
#pragma unroll
  for (int r = 0; r < 8; ++r) {
    int m = (lane < 16) ? r : (8 + r);
    int row = r0 + m;
    float v = acc[r] + bv;
    if (!transposed)
      y[((size_t)b * KB_N + row) * Oc + o] = (__bf16)v;
    else
      y[((size_t)b * Oc + o) * KB_N + row] = (__bf16)v;
  }
}

// ---------------------------------------------------------------------------
// Fused attention with async-LDS double buffering.
// Block = 8 waves = 128 query rows; each wave owns 16 rows.
// Per j-step of 32 keys:
//   (all 256 threads) async-stage K tile (32x64) and V tile (256x32) to LDS
//   S(16x32) = Q x K^T   (4 wmma, B-operands from LDS)
//   P = exp(clip(S/8))   (clip bounds logits -> no online max needed)
//   O(16x256) += P x V^T (16 wmma, B-operands from LDS)
// ---------------------------------------------------------------------------
#define KROW 72   // LDS row stride (bf16) for K tile: 144B, conflict-free
#define VROW 40   // LDS row stride (bf16) for V tile:  80B, conflict-free

__global__ __launch_bounds__(256) void sam_attn(
    const __bf16* __restrict__ qb, const __bf16* __restrict__ kb,
    const __bf16* __restrict__ vb, const float* __restrict__ x,
    const float* __restrict__ gamma, float* __restrict__ out) {
  __shared__ __align__(16) __bf16 sK[2][32 * KROW];   // [j][ch] padded
  __shared__ __align__(16) __bf16 sV[2][256 * VROW];  // [c][j]  padded
  __shared__ __align__(16) __bf16 sP[8][16 * 40];     // per-wave P transpose

  const int b = blockIdx.y;
  const int tid = threadIdx.x;
  const int wave = tid >> 5;
  const int lane = tid & 31;
  const int half = lane >> 4;
  const int lr = lane & 15;
  const int row0 = blockIdx.x * 128 + wave * 16;

  const __bf16* kbase = kb + (size_t)b * KB_N * KB_D;  // [n][64]
  const __bf16* vbase = vb + (size_t)b * KB_C * KB_N;  // [c][4096]

  // ---- persistent Q fragments (16 rows x 64 ch) ----
  const __bf16* qrow = qb + ((size_t)b * KB_N + row0 + lr) * KB_D;
  v16bf qa0 = load16(qrow + half * 8, 16);
  v16bf qa1 = load16(qrow + 32 + half * 8, 16);

  v8f oacc[16];
#pragma unroll
  for (int i = 0; i < 16; ++i) oacc[i] = (v8f){};
  v8f racc = {};

  // ---- async prefetch of one (K,V) tile pair: 5 b128 per thread ----
  const int kj = tid >> 3, kc = tid & 7;  // K: row j=tid/8, 16B chunk tid%8
  auto prefetch = [&](int buf, int jt) {
    async_b128(lds_off32(&sK[buf][kj * KROW + kc * 8]),
               kbase + ((size_t)(jt + kj)) * KB_D + kc * 8);
#pragma unroll
    for (int ck = 0; ck < 4; ++ck) {
      async_b128(lds_off32(&sV[buf][tid * VROW + ck * 8]),
                 vbase + (size_t)tid * KB_N + jt + ck * 8);
    }
  };

  prefetch(0, 0);

  const int nIter = KB_N / 32;
  for (int it = 0; it < nIter; ++it) {
    const int cur = it & 1;
    if (it + 1 < nIter) {
      prefetch(cur ^ 1, (it + 1) * 32);
      asm volatile("s_wait_asynccnt 0x5" ::: "memory");  // batch `it` done
    } else {
      asm volatile("s_wait_asynccnt 0x0" ::: "memory");
    }
    __syncthreads();  // tiles visible to all waves

    // ---- scores + exp for two j-groups of 16 ----
#pragma unroll
    for (int g = 0; g < 2; ++g) {
      const __bf16* kcol = &sK[cur][(g * 16 + lr) * KROW];
      v16bf b0 = load16(kcol + half * 8, 16);       // channels 0..31
      v16bf b1 = load16(kcol + 32 + half * 8, 16);  // channels 32..63
      v8f s = {};
      s = wmma_bf16(qa0, b0, s);
      s = wmma_bf16(qa1, b1, s);
#pragma unroll
      for (int r = 0; r < 8; ++r) {
        float t = s[r] * 0.125f;              // / sqrt(64)
        t = fminf(5.0f, fmaxf(-5.0f, t));     // clip
        t = __expf(t);                         // bounded -> no online max
        racc[r] += t;
        int m = (lane < 16) ? r : (8 + r);
        sP[wave][m * 40 + g * 16 + lr] = (__bf16)t;
      }
    }
    // ---- P as A-fragment, O += P x V ----
    v16bf pa = load16(&sP[wave][lr * 40 + half * 8], 16);
#pragma unroll
    for (int cc = 0; cc < 16; ++cc) {
      const __bf16* vrow = &sV[cur][(cc * 16 + lr) * VROW + half * 8];
      v16bf bvf = load16(vrow, 16);
      oacc[cc] = wmma_bf16(pa, bvf, oacc[cc]);
    }
    __syncthreads();  // everyone done with buffer `cur` before it refills
  }

  // ---- row-sum butterfly within each 16-lane half; racc[r] ends up holding
  // the full sum for exactly the row its C-fragment slot maps to ----
#pragma unroll
  for (int off = 1; off < 16; off <<= 1) {
#pragma unroll
    for (int r = 0; r < 8; ++r) racc[r] += __shfl_xor(racc[r], off, 32);
  }

  const float g = gamma[0];
#pragma unroll
  for (int cc = 0; cc < 16; ++cc) {
    int c = cc * 16 + lr;
#pragma unroll
    for (int r = 0; r < 8; ++r) {
      int m = (lane < 16) ? r : (8 + r);
      size_t idx = ((size_t)b * KB_C + c) * KB_N + (row0 + m);
      out[idx] = g * (oacc[cc][r] / racc[r]) + x[idx];
    }
  }
}

// ---------------------------------------------------------------------------
extern "C" void kernel_launch(void* const* d_in, const int* in_sizes, int n_in,
                              void* d_out, int out_size, void* d_ws,
                              size_t ws_size, hipStream_t stream) {
  (void)in_sizes; (void)n_in; (void)out_size; (void)ws_size;

  const float* x     = (const float*)d_in[0];
  const float* Wq    = (const float*)d_in[1];
  const float* bq    = (const float*)d_in[2];
  const float* Wk    = (const float*)d_in[3];
  const float* bk    = (const float*)d_in[4];
  const float* Wv    = (const float*)d_in[5];
  const float* bv    = (const float*)d_in[6];
  const float* gamma = (const float*)d_in[7];
  float* out = (float*)d_out;

  // workspace layout (bytes)
  char* ws = (char*)d_ws;
  __bf16* xb = (__bf16*)(ws);                              // 16 MiB  [b][n][c]
  __bf16* wq = (__bf16*)(ws + 16777216);                   // 32 KiB  [o][c]
  __bf16* wk = (__bf16*)(ws + 16777216 + 32768);           // 32 KiB
  __bf16* wv = (__bf16*)(ws + 16777216 + 65536);           // 128 KiB
  __bf16* qb = (__bf16*)(ws + 16777216 + 65536 + 131072);  // 4 MiB   [b][n][64]
  __bf16* kb = (__bf16*)(ws + 16777216 + 65536 + 131072 + 4194304);   // [b][n][64]
  __bf16* vb = (__bf16*)(ws + 16777216 + 65536 + 131072 + 8388608);   // [b][c][n]

  // 1) converts
  sam_convert_transpose_x<<<(KB_B * KB_N * KB_C) / 256, 256, 0, stream>>>(x, xb);
  sam_convert_bf16<<<(KB_D * KB_C + 255) / 256, 256, 0, stream>>>(Wq, wq, KB_D * KB_C);
  sam_convert_bf16<<<(KB_D * KB_C + 255) / 256, 256, 0, stream>>>(Wk, wk, KB_D * KB_C);
  sam_convert_bf16<<<(KB_C * KB_C + 255) / 256, 256, 0, stream>>>(Wv, wv, KB_C * KB_C);

  // 2) projections (WMMA GEMMs)
  sam_proj_gemm<<<dim3((KB_N / 128) * (KB_D / 16), KB_B), 256, 0, stream>>>(
      xb, wq, bq, qb, KB_D, 0);
  sam_proj_gemm<<<dim3((KB_N / 128) * (KB_D / 16), KB_B), 256, 0, stream>>>(
      xb, wk, bk, kb, KB_D, 0);
  sam_proj_gemm<<<dim3((KB_N / 128) * (KB_C / 16), KB_B), 256, 0, stream>>>(
      xb, wv, bv, vb, KB_C, 1);

  // 3) fused attention + residual
  sam_attn<<<dim3(KB_N / 128, KB_B), 256, 0, stream>>>(qb, kb, vb, x, gamma, out);
}
